// Decoder_45165876085035
// MI455X (gfx1250) — matrix operand order repaired and checked by reading, not compiled, not run on hardware
//
#include <hip/hip_runtime.h>
#include <math.h>

// ---------------------------------------------------------------------------
// Tacotron2 decoder for MI455X (gfx1250, wave32, WMMA).
// Strategy:
//   * one-time prep kernels: f32->bf16 weight conversion (L2-resident ~36MB),
//     prenet over all 400 frames, processed_memory GEMM, state zeroing
//   * ONE persistent kernel runs the 400-step scan with grid barriers
//     (L2 atomics) between the 9 stages of each step
//   * LSTM gate GEMMs use v_wmma_f32_16x16x32_bf16: 512 16x16 output tiles,
//     one wave per tile, A/B fetched coalesced straight from L2
//   * attention internals (conv/tanh/softmax/context) stay f32 VALU + LDS
// ---------------------------------------------------------------------------

typedef __attribute__((ext_vector_type(16))) __bf16 v16bf;
typedef __attribute__((ext_vector_type(8)))  __bf16 v8bf;
typedef __attribute__((ext_vector_type(8)))  float  v8f;

constexpr int B_    = 32;
constexpr int TXT   = 256;
constexpr int MELT  = 400;
constexpr int ENC   = 512;
constexpr int NMEL  = 80;
constexpr int PRE   = 256;
constexpr int RNN   = 1024;
constexpr int G4    = 4096;   // 4*RNN
constexpr int ADIM  = 128;
constexpr int NF    = 32;
constexpr int KS_   = 31;
constexpr float NEG = -1e9f;

constexpr int PBLK = 64;            // persistent blocks
constexpr int TPB  = 256;           // threads / block (8 waves)
constexpr int GSZ  = PBLK * TPB;    // 16384 threads
constexpr int NWAV = PBLK * 8;      // 512 waves

// ---- workspace layout (bytes; every offset is 256B-aligned) ---------------
constexpr size_t O_WIA = 0;                                   // 4096x768  bf16
constexpr size_t O_WHA = O_WIA + (size_t)G4 * 768 * 2;        // 4096x1024 bf16
constexpr size_t O_WID = O_WHA + (size_t)G4 * 1024 * 2;       // 4096x1536 bf16
constexpr size_t O_WHD = O_WID + (size_t)G4 * 1536 * 2;       // 4096x1024 bf16
constexpr size_t O_PN  = O_WHD + (size_t)G4 * 1024 * 2;       // 400x32x256 bf16
constexpr size_t O_X1  = O_PN  + (size_t)MELT * B_ * PRE * 2; // prenet tmp f32
constexpr size_t O_PM  = O_X1  + (size_t)MELT * B_ * PRE * 4; // 32x256x128 f32
constexpr size_t O_GT  = O_PM  + (size_t)B_ * TXT * ADIM * 4; // gates 32x4096
constexpr size_t O_HA  = O_GT  + (size_t)B_ * G4 * 4;
constexpr size_t O_CA  = O_HA  + (size_t)B_ * RNN * 4;
constexpr size_t O_HD  = O_CA  + (size_t)B_ * RNN * 4;
constexpr size_t O_CD  = O_HD  + (size_t)B_ * RNN * 4;
constexpr size_t O_HAB = O_CD  + (size_t)B_ * RNN * 4;        // bf16
constexpr size_t O_HDB = O_HAB + (size_t)B_ * RNN * 2;        // bf16
constexpr size_t O_CTX = O_HDB + (size_t)B_ * RNN * 2;        // f32
constexpr size_t O_CTXB= O_CTX + (size_t)B_ * ENC * 4;        // bf16
constexpr size_t O_AW  = O_CTXB+ (size_t)B_ * ENC * 2;
constexpr size_t O_AWC = O_AW  + (size_t)B_ * TXT * 4;
constexpr size_t O_E   = O_AWC + (size_t)B_ * TXT * 4;
constexpr size_t O_Q   = O_E   + (size_t)B_ * TXT * 4;        // 32x128 f32
constexpr size_t O_BAR = O_Q   + (size_t)B_ * ADIM * 4;       // barrier state

struct Params {
  const float* memory; const float* dec_in; const int* mlen;
  const float* pW1; const float* pW2;
  const float* WiA; const float* WhA; const float* bA;
  const float* convW; const float* ldW; const float* qW; const float* mW; const float* vW;
  const float* WiD; const float* WhD; const float* bD;
  const float* projW; const float* projB; const float* gateW; const float* gateB;
  float* out_mel; float* out_gate; float* out_aln;
  char* ws;
};

#define WSF(off)  ((float*)(p.ws + (off)))
#define WSU(off)  ((unsigned short*)(p.ws + (off)))
#define WSBF(off) ((const __bf16*)(p.ws + (off)))

__device__ __forceinline__ unsigned short f2bf(float x) {   // RNE f32->bf16
  unsigned u = __float_as_uint(x);
  unsigned r = u + 0x7fffu + ((u >> 16) & 1u);
  return (unsigned short)(r >> 16);
}
__device__ __forceinline__ float sigm(float x) { return 1.f / (1.f + expf(-x)); }

// ---- grid-wide split barrier via L2 atomics -------------------------------
__device__ __forceinline__ void gridbar(unsigned* bar) {
  __syncthreads();
  if (threadIdx.x == 0) {
    __threadfence();
    volatile unsigned* gen = bar + 64;            // separate cacheline
    unsigned g = *gen;
    if (atomicAdd(bar, 1u) == gridDim.x - 1) {
      bar[0] = 0;
      __threadfence();
      *gen = g + 1;
    } else {
      while (*gen == g) __builtin_amdgcn_s_sleep(2);
    }
    __threadfence();
  }
  __syncthreads();
}

// ---- WMMA K-loop over one bf16 segment ------------------------------------
// A: 16xK tile, row-major, lane m = lane&15; per-lane halves are 16
//    contiguous halves at k+g*16 (same k order as B).
// B[k][n] = W[n][k]: lane n = lane&15 reads 16 contiguous halves of W row n
//    at k+g*16 (contiguous k order, same as A).
template <int KLEN>
__device__ __forceinline__ v8f kloop(v8f acc, const __bf16* __restrict__ arow,
                                     const __bf16* __restrict__ wrow, int g) {
#pragma unroll 8
  for (int kb = 0; kb < KLEN; kb += 32) {
    union { v16bf v; v8bf h[2]; } a;
    a.h[0] = *(const v8bf*)(arow + kb + g * 16);
    a.h[1] = *(const v8bf*)(arow + kb + g * 16 + 8);
    v16bf bm = *(const v16bf*)(wrow + kb + g * 16);
    acc = __builtin_amdgcn_wmma_f32_16x16x32_bf16(false, a.v, false, bm,
                                                  (short)0, acc, false, false);
  }
  return acc;
}

__device__ __forceinline__ void lstm_pw(const float* __restrict__ gates,
                                        const float* __restrict__ bias,
                                        float* h, float* c, unsigned short* hbf,
                                        int gtid) {
  for (int i = gtid; i < B_ * RNN; i += GSZ) {
    int b = i >> 10, n = i & 1023;
    const float* gr = gates + b * G4;
    float ig = gr[n]          + bias[n];
    float fg = gr[1024 + n]   + bias[1024 + n];
    float gg = gr[2048 + n]   + bias[2048 + n];
    float og = gr[3072 + n]   + bias[3072 + n];
    float cn = sigm(fg) * c[i] + sigm(ig) * tanhf(gg);
    float hn = sigm(og) * tanhf(cn);
    c[i] = cn; h[i] = hn; hbf[i] = f2bf(hn);
  }
}

// ============================ prep kernels =================================

__global__ void __launch_bounds__(256) k_prep(Params p) {
  size_t gid = (size_t)blockIdx.x * blockDim.x + threadIdx.x;
  size_t gsz = (size_t)gridDim.x * blockDim.x;
  unsigned short* wiA = WSU(O_WIA); unsigned short* whA = WSU(O_WHA);
  unsigned short* wiD = WSU(O_WID); unsigned short* whD = WSU(O_WHD);
  for (size_t i = gid; i < (size_t)G4 * 768;  i += gsz) wiA[i] = f2bf(p.WiA[i]);
  for (size_t i = gid; i < (size_t)G4 * 1024; i += gsz) whA[i] = f2bf(p.WhA[i]);
  for (size_t i = gid; i < (size_t)G4 * 1536; i += gsz) wiD[i] = f2bf(p.WiD[i]);
  for (size_t i = gid; i < (size_t)G4 * 1024; i += gsz) whD[i] = f2bf(p.WhD[i]);
  float* hA = WSF(O_HA); float* cA = WSF(O_CA);
  float* hD = WSF(O_HD); float* cD = WSF(O_CD);
  unsigned short* hAb = WSU(O_HAB); unsigned short* hDb = WSU(O_HDB);
  for (size_t i = gid; i < (size_t)B_ * RNN; i += gsz) {
    hA[i] = 0.f; cA[i] = 0.f; hD[i] = 0.f; cD[i] = 0.f; hAb[i] = 0; hDb[i] = 0;
  }
  float* ctx = WSF(O_CTX); unsigned short* ctxb = WSU(O_CTXB);
  for (size_t i = gid; i < (size_t)B_ * ENC; i += gsz) { ctx[i] = 0.f; ctxb[i] = 0; }
  float* aw = WSF(O_AW); float* awc = WSF(O_AWC);
  for (size_t i = gid; i < (size_t)B_ * TXT; i += gsz) { aw[i] = 0.f; awc[i] = 0.f; }
  if (gid == 0) { unsigned* bar = (unsigned*)(p.ws + O_BAR); bar[0] = 0; bar[64] = 0; }
}

// prenet layer 1: x1[s*32+b][q] = relu(frames(s,b,:) . W1[q,:])
__global__ void __launch_bounds__(256) k_prenet1(Params p) {
  float* x1 = WSF(O_X1);
  size_t gid = (size_t)blockIdx.x * blockDim.x + threadIdx.x;
  size_t gsz = (size_t)gridDim.x * blockDim.x;
  for (size_t i = gid; i < (size_t)MELT * B_ * PRE; i += gsz) {
    int row = (int)(i >> 8), q = (int)(i & 255);
    int s = row >> 5, b = row & 31;
    float acc = 0.f;
    if (s > 0) {
      const float* f = p.dec_in + (size_t)b * NMEL * MELT + (s - 1); // stride MELT
      const float* w = p.pW1 + q * NMEL;
      for (int m = 0; m < NMEL; ++m) acc += f[(size_t)m * MELT] * w[m];
    }
    x1[i] = fmaxf(acc, 0.f);
  }
}

// prenet layer 2 -> bf16: pn[row][q] = relu(x1[row,:] . W2[q,:])
__global__ void __launch_bounds__(256) k_prenet2(Params p) {
  const float* x1 = WSF(O_X1);
  unsigned short* pn = WSU(O_PN);
  size_t gid = (size_t)blockIdx.x * blockDim.x + threadIdx.x;
  size_t gsz = (size_t)gridDim.x * blockDim.x;
  for (size_t i = gid; i < (size_t)MELT * B_ * PRE; i += gsz) {
    int row = (int)(i >> 8), q = (int)(i & 255);
    const float* xr = x1 + (size_t)row * PRE;
    const float* w  = p.pW2 + q * PRE;
    float acc = 0.f;
    for (int k = 0; k < PRE; ++k) acc += xr[k] * w[k];
    pn[i] = f2bf(fmaxf(acc, 0.f));
  }
}

// processed_memory: pm[b,t,a] = memory[b,t,:] . mW[a,:]
__global__ void __launch_bounds__(256) k_procmem(Params p) {
  float* pm = WSF(O_PM);
  size_t gid = (size_t)blockIdx.x * blockDim.x + threadIdx.x;
  size_t gsz = (size_t)gridDim.x * blockDim.x;
  for (size_t i = gid; i < (size_t)B_ * TXT * ADIM; i += gsz) {
    int bt = (int)(i >> 7), a = (int)(i & 127);
    const float* mr = p.memory + (size_t)bt * ENC;
    const float* w  = p.mW + a * ENC;
    float acc = 0.f;
    for (int e = 0; e < ENC; ++e) acc += mr[e] * w[e];
    pm[i] = acc;
  }
}

// ========================= persistent decoder ==============================

__global__ void __launch_bounds__(TPB) k_decoder(Params p) {
  __shared__ float s_ldW[ADIM * NF];      // loc_dense  [a][f]
  __shared__ float s_convW[NF * 2 * KS_]; // loc_conv   [f][c][k]
  __shared__ float s_v[ADIM];
  __shared__ float s_red[TPB];

  const int tid  = threadIdx.x;
  const int gtid = blockIdx.x * TPB + tid;
  const int lane = tid & 31;
  const int gwid = blockIdx.x * 8 + (tid >> 5);

  for (int i = tid; i < ADIM * NF;      i += TPB) s_ldW[i]   = p.ldW[i];
  for (int i = tid; i < NF * 2 * KS_;   i += TPB) s_convW[i] = p.convW[i];
  for (int i = tid; i < ADIM;           i += TPB) s_v[i]     = p.vW[i];
  __syncthreads();

  float* gates = WSF(O_GT);
  float* hA = WSF(O_HA); float* cA = WSF(O_CA);
  float* hD = WSF(O_HD); float* cD = WSF(O_CD);
  unsigned short* hAbw = WSU(O_HAB); unsigned short* hDbw = WSU(O_HDB);
  const __bf16* hAb = WSBF(O_HAB); const __bf16* hDb = WSBF(O_HDB);
  float* ctx = WSF(O_CTX); unsigned short* ctxbw = WSU(O_CTXB);
  const __bf16* ctxb = WSBF(O_CTXB);
  float* aw = WSF(O_AW); float* awc = WSF(O_AWC);
  float* ebuf = WSF(O_E); float* qv = WSF(O_Q);
  const float* pm = WSF(O_PM);
  const __bf16* pn  = WSBF(O_PN);
  const __bf16* wiA = WSBF(O_WIA); const __bf16* whA = WSBF(O_WHA);
  const __bf16* wiD = WSBF(O_WID); const __bf16* whD = WSBF(O_WHD);
  unsigned* bar = (unsigned*)(p.ws + O_BAR);

  for (int s = 0; s < MELT; ++s) {
    // ---- P1: attention-LSTM gates = [pn_s|ctx] @ WiA^T + hA @ WhA^T (WMMA)
    if (gwid < NWAV && gwid < 512) {
      const int nt = gwid >> 1, mt = gwid & 1;
      const int g = lane >> 4, mr = lane & 15;
      const int am = mt * 16 + mr;
      v8f acc = {};
      acc = kloop<256> (acc, pn + (size_t)s * (B_ * PRE) + am * PRE,
                        wiA + (size_t)(nt * 16 + mr) * 768, g);
      acc = kloop<512> (acc, ctxb + am * ENC,
                        wiA + (size_t)(nt * 16 + mr) * 768 + 256, g);
      acc = kloop<1024>(acc, hAb + am * RNN,
                        whA + (size_t)(nt * 16 + mr) * 1024, g);
      const int n = nt * 16 + mr, mb = mt * 16 + g * 8;
#pragma unroll
      for (int r = 0; r < 8; ++r) gates[(size_t)(mb + r) * G4 + n] = acc[r];
    }
    gridbar(bar);

    // ---- P2: attention-LSTM pointwise -> hA, cA, hAb
    lstm_pw(gates, p.bA, hA, cA, hAbw, gtid);
    gridbar(bar);

    // ---- P3: q[b,a] = hA[b,:] . qW[a,:]
    if (gtid < B_ * ADIM) {
      int b = gtid >> 7, a = gtid & 127;
      const float* h = hA + b * RNN;
      const float* w = p.qW + a * RNN;
      float acc = 0.f;
      for (int k = 0; k < RNN; ++k) acc += h[k] * w[k];
      qv[gtid] = acc;
    }
    gridbar(bar);

    // ---- P4: energies; wave per (b,t): conv (lane=filter) + tanh dot v
    for (int it = 0; it < (B_ * TXT) / NWAV; ++it) {
      int item = gwid + it * NWAV;           // uniform per wave
      if (item < B_ * TXT) {
        int b = item >> 8, t = item & 255;
        const float* wr = &s_convW[lane * 62];
        const float* awr = aw + b * TXT;
        const float* awcr = awc + b * TXT;
        float cf = 0.f;
#pragma unroll
        for (int k = 0; k < KS_; ++k) {
          int tt = t + k - 15;
          if (tt >= 0 && tt < TXT) cf += wr[k] * awr[tt] + wr[KS_ + k] * awcr[tt];
        }
        float accj[4] = {0.f, 0.f, 0.f, 0.f};
        for (int f = 0; f < NF; ++f) {
          float cv = __shfl(cf, f, 32);
#pragma unroll
          for (int j = 0; j < 4; ++j) accj[j] += s_ldW[(lane + 32 * j) * NF + f] * cv;
        }
        float part = 0.f;
#pragma unroll
        for (int j = 0; j < 4; ++j) {
          int a = lane + 32 * j;
          float sarg = qv[b * ADIM + a] + pm[(size_t)item * ADIM + a] + accj[j];
          part += s_v[a] * tanhf(sarg);
        }
        for (int off = 16; off; off >>= 1) part += __shfl_xor(part, off, 32);
        if (lane == 0) ebuf[item] = (t < p.mlen[b]) ? part : NEG;
      }
    }
    gridbar(bar);

    // ---- P5: softmax per batch row (block b), update aw/awc, emit alignments
    if (blockIdx.x < B_) {
      int b = blockIdx.x, t = tid;
      float ev = ebuf[b * TXT + t];
      s_red[t] = ev; __syncthreads();
      for (int st = 128; st; st >>= 1) {
        if (t < st) s_red[t] = fmaxf(s_red[t], s_red[t + st]);
        __syncthreads();
      }
      float mx = s_red[0]; __syncthreads();
      float ex = expf(ev - mx);
      s_red[t] = ex; __syncthreads();
      for (int st = 128; st; st >>= 1) {
        if (t < st) s_red[t] += s_red[t + st];
        __syncthreads();
      }
      float a = ex / s_red[0];
      aw[b * TXT + t] = a;
      awc[b * TXT + t] += a;
      p.out_aln[((size_t)b * MELT + s) * TXT + t] = a;
    }
    gridbar(bar);

    // ---- P6: context ctx[b,e] = sum_t aw[b,t] * memory[b,t,e]
    {
      int b = gtid >> 9, e0 = gtid & 511;
      const float* awr = aw + b * TXT;
      const float* mem = p.memory + (size_t)b * TXT * ENC + e0;
      float acc = 0.f;
      for (int t = 0; t < TXT; ++t) acc += awr[t] * mem[(size_t)t * ENC];
      ctx[gtid] = acc;
      ctxbw[gtid] = f2bf(acc);
    }
    gridbar(bar);

    // ---- P7: decoder-LSTM gates = [hA|ctx] @ WiD^T + hD @ WhD^T (WMMA)
    if (gwid < 512) {
      const int nt = gwid >> 1, mt = gwid & 1;
      const int g = lane >> 4, mr = lane & 15;
      const int am = mt * 16 + mr;
      v8f acc = {};
      acc = kloop<1024>(acc, hAb + am * RNN,
                        wiD + (size_t)(nt * 16 + mr) * 1536, g);
      acc = kloop<512> (acc, ctxb + am * ENC,
                        wiD + (size_t)(nt * 16 + mr) * 1536 + 1024, g);
      acc = kloop<1024>(acc, hDb + am * RNN,
                        whD + (size_t)(nt * 16 + mr) * 1024, g);
      const int n = nt * 16 + mr, mb = mt * 16 + g * 8;
#pragma unroll
      for (int r = 0; r < 8; ++r) gates[(size_t)(mb + r) * G4 + n] = acc[r];
    }
    gridbar(bar);

    // ---- P8: decoder-LSTM pointwise -> hD, cD, hDb
    lstm_pw(gates, p.bD, hD, cD, hDbw, gtid);
    gridbar(bar);

    // ---- P9: projection + gate outputs
    if (gtid < B_ * NMEL + B_) {
      if (gtid < B_ * NMEL) {
        int b = gtid / NMEL, m = gtid % NMEL;
        const float* w = p.projW + m * (RNN + ENC);
        const float* h = hD + b * RNN;
        const float* cx = ctx + b * ENC;
        float acc = p.projB[m];
        for (int k = 0; k < RNN; ++k) acc += h[k] * w[k];
        for (int k = 0; k < ENC; ++k) acc += cx[k] * w[RNN + k];
        p.out_mel[(size_t)gtid * MELT + s] = acc;   // (b*80+m)*400+s
      } else {
        int b = gtid - B_ * NMEL;
        const float* h = hD + b * RNN;
        const float* cx = ctx + b * ENC;
        float acc = p.gateB[0];
        for (int k = 0; k < RNN; ++k) acc += h[k] * p.gateW[k];
        for (int k = 0; k < ENC; ++k) acc += cx[k] * p.gateW[RNN + k];
        p.out_gate[(size_t)b * MELT + s] = acc;
      }
    }
    gridbar(bar);
  }
}

// ================================ launch ===================================

extern "C" void kernel_launch(void* const* d_in, const int* in_sizes, int n_in,
                              void* d_out, int out_size, void* d_ws, size_t ws_size,
                              hipStream_t stream) {
  (void)in_sizes; (void)n_in; (void)out_size; (void)ws_size;
  Params p;
  p.memory = (const float*)d_in[0];
  p.dec_in = (const float*)d_in[1];
  p.mlen   = (const int*)d_in[2];
  p.pW1 = (const float*)d_in[3];  p.pW2 = (const float*)d_in[4];
  p.WiA = (const float*)d_in[5];  p.WhA = (const float*)d_in[6];  p.bA = (const float*)d_in[7];
  p.convW = (const float*)d_in[8]; p.ldW = (const float*)d_in[9];
  p.qW = (const float*)d_in[10];   p.mW  = (const float*)d_in[11]; p.vW = (const float*)d_in[12];
  p.WiD = (const float*)d_in[13];  p.WhD = (const float*)d_in[14]; p.bD = (const float*)d_in[15];
  p.projW = (const float*)d_in[16]; p.projB = (const float*)d_in[17];
  p.gateW = (const float*)d_in[18]; p.gateB = (const float*)d_in[19];
  float* out = (float*)d_out;
  p.out_mel  = out;
  p.out_gate = out + (size_t)B_ * NMEL * MELT;
  p.out_aln  = out + (size_t)B_ * NMEL * MELT + (size_t)B_ * MELT;
  p.ws = (char*)d_ws;

  hipLaunchKernelGGL(k_prep,    dim3(512), dim3(256), 0, stream, p);
  hipLaunchKernelGGL(k_prenet1, dim3(512), dim3(256), 0, stream, p);
  hipLaunchKernelGGL(k_prenet2, dim3(512), dim3(256), 0, stream, p);
  hipLaunchKernelGGL(k_procmem, dim3(512), dim3(256), 0, stream, p);
  hipLaunchKernelGGL(k_decoder, dim3(PBLK), dim3(TPB), 0, stream, p);
}
